// VectorQuantizerEMA_7275674599501
// MI455X (gfx1250) — matrix operand order repaired and checked by reading, not compiled, not run on hardware
//
#include <hip/hip_runtime.h>
#include <hip/hip_bf16.h>

// ---- types ----
typedef __attribute__((ext_vector_type(16))) __bf16        v16bf;
typedef __attribute__((ext_vector_type(8)))  float         v8f;
typedef __attribute__((ext_vector_type(4)))  unsigned int  u32x4;
typedef __attribute__((ext_vector_type(2)))  unsigned int  u32x2;
typedef int v4i __attribute__((vector_size(16)));

union FragBF {
    u32x4 u[2];   // 2 x 16B = 32B
    v16bf v;      // 16 bf16
};
union BF4 {
    __bf16 h[4];
    u32x2  u;
};

#define DDIM   512
#define KCODES 4096
#define NTOK   16384          // B*L = 8*2048
#define MTILE  64
#define NTILE  32
#define NTILES (KCODES / NTILE)
#define DECAYF 0.99f
#define EPSF   1e-5f

// ---- CDNA5 async global->LDS helpers (guarded; fallback = sync copy) ----
typedef __attribute__((address_space(1))) v4i* gv4i_p;   // global (AS1)
typedef __attribute__((address_space(3))) v4i* lv4i_p;   // LDS (AS3)

__device__ __forceinline__ void async_b128(const void* g, void* l) {
#if __has_builtin(__builtin_amdgcn_global_load_async_to_lds_b128)
    __builtin_amdgcn_global_load_async_to_lds_b128(
        (gv4i_p)(unsigned long long)g,
        (lv4i_p)(unsigned int)(unsigned long long)l,
        0, 0);
#else
    *(u32x4*)l = *(const u32x4*)g;
#endif
}

template <int N> __device__ __forceinline__ void wait_async() {
#if __has_builtin(__builtin_amdgcn_s_wait_asynccnt)
    __builtin_amdgcn_s_wait_asynccnt(N);
#else
    asm volatile("s_wait_asynccnt %0" :: "i"(N));
#endif
}

// ---------------- zero workspace ----------------
__global__ void zero_ws_kernel(float* p, int n) {
    int i = blockIdx.x * blockDim.x + threadIdx.x;
    if (i < n) p[i] = 0.0f;
}

// ---------------- fp32 -> bf16 copy (vectorized, 4 elems/thread) ----------------
__global__ void cvt_bf16_kernel(const float* __restrict__ src, __bf16* __restrict__ dst, int n) {
    int base = (blockIdx.x * blockDim.x + threadIdx.x) * 4;
    if (base >= n) return;
    float4 f = *(const float4*)(src + base);
    BF4 h;
    h.h[0] = (__bf16)f.x; h.h[1] = (__bf16)f.y;
    h.h[2] = (__bf16)f.z; h.h[3] = (__bf16)f.w;
    *(u32x2*)(dst + base) = h.u;
}

// ---------------- c2[k] = sum codebook[k,:]^2 (wave per row) ----------------
__global__ void c2_kernel(const float* __restrict__ cb, float* __restrict__ c2) {
    const int lane = threadIdx.x & 31;
    const int wave = threadIdx.x >> 5;
    const int row  = blockIdx.x * 8 + wave;
    if (row >= KCODES) return;
    const float* r = cb + (size_t)row * DDIM;
    float s = 0.0f;
    for (int d = lane * 4; d < DDIM; d += 128) {
        float4 v = *(const float4*)(r + d);
        s += v.x * v.x + v.y * v.y + v.z * v.z + v.w * v.w;
    }
    #pragma unroll
    for (int off = 16; off > 0; off >>= 1) s += __shfl_xor(s, off, 32);
    if (lane == 0) c2[row] = s;
}

// ---------------- fused BF16 WMMA distance GEMM + argmin ----------------
// Block: 256 threads = 8 waves, owns MTILE=64 token rows.
// Waves: 4 row-groups x 2 col-groups of 16. B tiles (NTILE=32 codebook rows,
// already bf16 in global) are async-copied global->LDS, double buffered,
// pipelined with s_wait_asynccnt. 16 WMMA k-steps cover D=512, split into two
// independent accumulator chains for XDL-pipe ILP.
__global__ __launch_bounds__(256) void argmin_kernel(
        const float* __restrict__ x, const __bf16* __restrict__ cbh,
        const float* __restrict__ c2v, int* __restrict__ idxOut) {
    extern __shared__ char smem_raw[];
    __bf16* As  = (__bf16*)smem_raw;       // [MTILE][DDIM] (64 KB)
    __bf16* Bs0 = As + MTILE * DDIM;       // [NTILE][DDIM] (32 KB)
    __bf16* Bs1 = Bs0 + NTILE * DDIM;      // [NTILE][DDIM] (32 KB)

    const int tid   = threadIdx.x;
    const int mbase = blockIdx.x * MTILE;

    // Stage x tile into LDS as bf16 (float4 load -> 4x cvt -> b64 LDS store).
    for (int i = tid * 4; i < MTILE * DDIM; i += 256 * 4) {
        float4 f = *(const float4*)(x + (size_t)mbase * DDIM + i);
        BF4 h;
        h.h[0] = (__bf16)f.x; h.h[1] = (__bf16)f.y;
        h.h[2] = (__bf16)f.z; h.h[3] = (__bf16)f.w;
        *(u32x2*)(As + i) = h.u;
    }

    // Async-issue B tiles 0 and 1 (8 x b128 per thread per tile).
    auto issueTile = [&](int nt, __bf16* buf) {
        const __bf16* src = cbh + (size_t)nt * DDIM;
        #pragma unroll
        for (int j = 0; j < 8; ++j) {
            const int c = tid + 256 * j;          // b128 chunk id (8 bf16 each)
            async_b128(src + c * 8, buf + c * 8);
        }
    };
    issueTile(0, Bs0);
    issueTile(NTILE, Bs1);

    __syncthreads();   // A tile visible to all waves

    const int lane    = tid & 31;
    const int wave    = tid >> 5;
    const int wrow    = wave & 3;
    const int wcol    = wave >> 2;
    const int halfsel = lane >> 4;
    const int lcol    = lane & 15;

    // Hoist A fragments for the whole D=512 (16 k-steps x 8 VGPRs).
    // 16-bit A 16x32 layout: lanes 0-15 hold K {0..7,16..23}, lanes 16-31 hold
    // K {8..15,24..31} of row M=lane%16 -> two aligned 16B chunks per k-step.
    FragBF a[16];
    {
        const __bf16* ab = As + (size_t)(wrow * 16 + lcol) * DDIM + halfsel * 8;
        #pragma unroll
        for (int s = 0; s < 16; ++s) {
            a[s].u[0] = *(const u32x4*)(ab + 32 * s);
            a[s].u[1] = *(const u32x4*)(ab + 32 * s + 16);
        }
    }

    float bestV[8];
    int   bestI[8];
    #pragma unroll
    for (int r = 0; r < 8; ++r) { bestV[r] = 3.402823466e38f; bestI[r] = 0; }

    // 16-bit B 32x16 layout: column N = lane%16, lanes 0-15 hold K {0..7,16..23},
    // lanes 16-31 hold K {8..15,24..31} -> two aligned 16B chunks from the (contiguous)
    // codebook row in LDS.
    const int bbOff = (wcol * 16 + lcol) * DDIM + halfsel * 8;

    for (int it = 0; it < NTILES; ++it) {
        __bf16* bufc = (it & 1) ? Bs1 : Bs0;

        // Tile 'it' arrived when at most the next tile's 8 loads are pending.
        if (it + 1 < NTILES) wait_async<8>(); else wait_async<0>();
        __syncthreads();

        // Two independent accumulator chains (even/odd k-steps).
        v8f acc0 = {}, acc1 = {};
        const __bf16* bb = bufc + bbOff;
        #pragma unroll
        for (int s = 0; s < 16; s += 2) {
            FragBF b0, b1;
            b0.u[0] = *(const u32x4*)(bb + 32 * s);
            b0.u[1] = *(const u32x4*)(bb + 32 * s + 16);
            b1.u[0] = *(const u32x4*)(bb + 32 * (s + 1));
            b1.u[1] = *(const u32x4*)(bb + 32 * (s + 1) + 16);
            acc0 = __builtin_amdgcn_wmma_f32_16x16x32_bf16(
                false, a[s].v, false, b0.v, (short)0, acc0, false, false);
            acc1 = __builtin_amdgcn_wmma_f32_16x16x32_bf16(
                false, a[s + 1].v, false, b1.v, (short)0, acc1, false, false);
        }

        // C/D layout: VGPR r -> (M = r + 8*halfsel, N = lane%16).
        const int   nt  = it * NTILE;
        const int   col = nt + wcol * 16 + lcol;
        const float csq = c2v[col];
        #pragma unroll
        for (int r = 0; r < 8; ++r) {
            float sc = csq - 2.0f * (acc0[r] + acc1[r]);
            if (sc < bestV[r]) { bestV[r] = sc; bestI[r] = col; }
        }

        __syncthreads();                         // all waves done reading bufc
        if (it + 2 < NTILES) issueTile(nt + 2 * NTILE, bufc);
    }

    // Cross-lane argmin within each 16-lane half (xor masks 1..8 stay in-half).
    #pragma unroll
    for (int r = 0; r < 8; ++r) {
        float v = bestV[r]; int ix = bestI[r];
        #pragma unroll
        for (int off = 1; off < 16; off <<= 1) {
            float ov = __shfl_xor(v, off, 32);
            int   oi = __shfl_xor(ix, off, 32);
            if (ov < v || (ov == v && oi < ix)) { v = ov; ix = oi; }
        }
        bestV[r] = v; bestI[r] = ix;
    }
    if (lcol == 0) {
        const int rowbase = mbase + wrow * 16 + halfsel * 8;
        #pragma unroll
        for (int r = 0; r < 8; ++r) idxOut[rowbase + r] = bestI[r];
    }
}

// ---------------- scatter-add counts and dw (wave per token) ----------------
__global__ void scatter_kernel(const float* __restrict__ x, const int* __restrict__ idx,
                               float* __restrict__ counts, float* __restrict__ dw) {
    const int lane = threadIdx.x & 31;
    const int wave = threadIdx.x >> 5;
    const int t = blockIdx.x * 8 + wave;
    if (t >= NTOK) return;
    const int k = idx[t];
    const float* xr = x  + (size_t)t * DDIM;
    float*       dr = dw + (size_t)k * DDIM;
    for (int d = lane; d < DDIM; d += 32) atomicAdd(&dr[d], xr[d]);
    if (lane == 0) atomicAdd(&counts[k], 1.0f);
}

// ---------------- EMA cluster + N reduction ----------------
__global__ void ema_cluster_kernel(const float* __restrict__ hc, const float* __restrict__ counts,
                                   const float* __restrict__ countScalar,
                                   float* __restrict__ avg, float* __restrict__ Nacc) {
    int k = blockIdx.x * blockDim.x + threadIdx.x;
    if (k >= KCODES) return;
    float bias = 1.0f - powf(DECAYF, countScalar[0] + 1.0f);
    float a = (hc[k] * DECAYF + (1.0f - DECAYF) * counts[k]) / bias;
    avg[k] = a;
    atomicAdd(Nacc, a);
}

// ---------------- codebook_new ----------------
__global__ void codebook_new_kernel(const float* __restrict__ hdw, const float* __restrict__ dw,
                                    const float* __restrict__ avg, const float* __restrict__ Nacc,
                                    const float* __restrict__ countScalar, float* __restrict__ cbn) {
    int i = blockIdx.x * blockDim.x + threadIdx.x;
    if (i >= KCODES * DDIM) return;
    int k = i / DDIM;
    float bias = 1.0f - powf(DECAYF, countScalar[0] + 1.0f);
    float N  = Nacc[0];
    float cc = (avg[k] + EPSF) / (N + (float)KCODES * EPSF) * N;
    cbn[i] = ((hdw[i] * DECAYF + (1.0f - DECAYF) * dw[i]) / bias) / cc;
}

// ---------------- gather quantized + loss reduction (wave per token) ----------------
__global__ void gather_loss_kernel(const float* __restrict__ x, const int* __restrict__ idx,
                                   const float* __restrict__ cbn,
                                   float* __restrict__ quant, float* __restrict__ lossAcc) {
    const int lane = threadIdx.x & 31;
    const int wave = threadIdx.x >> 5;
    const int t = blockIdx.x * 8 + wave;
    if (t >= NTOK) return;
    const int k = idx[t];
    const float* xr = x   + (size_t)t * DDIM;
    const float* cr = cbn + (size_t)k * DDIM;
    float*       qr = quant + (size_t)t * DDIM;
    float s = 0.0f;
    for (int d = lane * 4; d < DDIM; d += 128) {
        float4 q  = *(const float4*)(cr + d);
        float4 xv = *(const float4*)(xr + d);
        *(float4*)(qr + d) = q;
        float d0 = xv.x - q.x, d1 = xv.y - q.y, d2 = xv.z - q.z, d3 = xv.w - q.w;
        s += d0 * d0 + d1 * d1 + d2 * d2 + d3 * d3;
    }
    #pragma unroll
    for (int off = 16; off > 0; off >>= 1) s += __shfl_xor(s, off, 32);
    if (lane == 0) atomicAdd(lossAcc, s);
}

// ---------------- finalize: loss scalar + indices into d_out ----------------
__global__ void finalize_kernel(const float* __restrict__ lossAcc, const int* __restrict__ idx,
                                float* __restrict__ out) {
    int i = blockIdx.x * blockDim.x + threadIdx.x;
    if (i == 0)
        out[(size_t)NTOK * DDIM] = 0.5f * lossAcc[0] / ((float)NTOK * (float)DDIM);
    if (i < NTOK)
        out[(size_t)NTOK * DDIM + 1 + i] = (float)idx[i];
}

extern "C" void kernel_launch(void* const* d_in, const int* in_sizes, int n_in,
                              void* d_out, int out_size, void* d_ws, size_t ws_size,
                              hipStream_t stream) {
    const float* x   = (const float*)d_in[0];   // (8,2048,512)
    const float* cb  = (const float*)d_in[1];   // (4096,512)
    const float* hc  = (const float*)d_in[2];   // (4096,)
    const float* hdw = (const float*)d_in[3];   // (4096,512)
    const float* cnt = (const float*)d_in[4];   // scalar
    float* out = (float*)d_out;

    // workspace layout
    float*  ws      = (float*)d_ws;
    int*    idx     = (int*)ws;                      // NTOK ints
    float*  counts  = ws + NTOK;                     // KCODES
    float*  dw      = counts + KCODES;               // KCODES*DDIM
    float*  avg     = dw + (size_t)KCODES * DDIM;    // KCODES
    float*  cbn     = avg + KCODES;                  // KCODES*DDIM
    float*  c2      = cbn + (size_t)KCODES * DDIM;   // KCODES
    float*  Nacc    = c2 + KCODES;                   // 1
    float*  lossAcc = Nacc + 1;                      // 1
    __bf16* cbh     = (__bf16*)(lossAcc + 1);        // KCODES*DDIM bf16 (4 MB)

    // zero counts..lossAcc
    const int nz = KCODES + KCODES * DDIM + KCODES + KCODES * DDIM + KCODES + 2;
    zero_ws_kernel<<<(nz + 255) / 256, 256, 0, stream>>>(counts, nz);

    // bf16 codebook copy (one pass; stays hot in 192MB L2 for the GEMM)
    cvt_bf16_kernel<<<(KCODES * DDIM / 4 + 255) / 256, 256, 0, stream>>>(cb, cbh, KCODES * DDIM);

    // codebook squared norms
    c2_kernel<<<KCODES / 8, 256, 0, stream>>>(cb, c2);

    // fused WMMA distance GEMM + argmin (A tile + double-buffered B tiles)
    const size_t ldsBytes = (size_t)(MTILE + 2 * NTILE) * DDIM * sizeof(__bf16); // 128 KB
    argmin_kernel<<<NTOK / MTILE, 256, ldsBytes, stream>>>(x, cbh, c2, idx);

    // scatter-add
    scatter_kernel<<<NTOK / 8, 256, 0, stream>>>(x, idx, counts, dw);

    // EMA
    ema_cluster_kernel<<<(KCODES + 255) / 256, 256, 0, stream>>>(hc, counts, cnt, avg, Nacc);
    codebook_new_kernel<<<(KCODES * DDIM + 255) / 256, 256, 0, stream>>>(hdw, dw, avg, Nacc, cnt, cbn);

    // gather + loss
    gather_loss_kernel<<<NTOK / 8, 256, 0, stream>>>(x, idx, cbn, out, lossAcc);

    // finalize outputs
    finalize_kernel<<<(NTOK + 255) / 256, 256, 0, stream>>>(lossAcc, idx, out);
}
